// CSACMRFusion_66597762892475
// MI455X (gfx1250) — compile-verified, matrix operand order checked by reading, not run on hardware
//
#include <hip/hip_runtime.h>
#include <hip/hip_bf16.h>
#include <stdint.h>

typedef __attribute__((ext_vector_type(16))) __bf16 v16bf;
typedef __attribute__((ext_vector_type(8)))  float  v8f;

union Frag { v16bf v; uint4 q[2]; };
union H8   { uint4 q; __bf16 h[8]; };

static constexpr int kH = 64, kW = 64, kHW = 4096;
static constexpr int kB = 8;
static constexpr int kNPix = kB * kHW;      // 32768 pixels
static constexpr int kMTiles = kNPix / 16;  // 2048 M-tiles of 16 x-contiguous pixels

// ---------------------------------------------------------------------------
// Weight packing: w[Co][Ci][3][3] fp32 -> wT[tap][Npad][Ci] bf16 (zero-pad co)
// ---------------------------------------------------------------------------
__global__ void pack_w_kernel(const float* __restrict__ w, __bf16* __restrict__ dst,
                              int Co, int Ci, int Npad)
{
    int idx = blockIdx.x * blockDim.x + threadIdx.x;
    int total = 9 * Npad * Ci;
    if (idx >= total) return;
    int t  = idx / (Npad * Ci);
    int r  = idx - t * (Npad * Ci);
    int co = r / Ci;
    int ci = r - co * Ci;
    float v = (co < Co) ? w[((size_t)co * Ci + ci) * 9 + t] : 0.f;
    dst[idx] = (__bf16)v;   // dst layout (t,co,ci) flattened == idx
}

// ---------------------------------------------------------------------------
// NCHW fp32 -> NHWC-slot bf16 (dstStride channels per pixel, at chOff)
// ---------------------------------------------------------------------------
__global__ void nchw_to_slot_kernel(const float* __restrict__ src, __bf16* __restrict__ dst,
                                    int C, int dstStride, int chOff)
{
    int idx = blockIdx.x * blockDim.x + threadIdx.x;
    if (idx >= kNPix * C) return;
    int c = idx / kNPix;
    int p = idx - c * kNPix;
    int b = p >> 12, rem = p & (kHW - 1);
    dst[(size_t)p * dstStride + chOff + c] = (__bf16)src[((size_t)b * C + c) * kHW + rem];
}

// ---------------------------------------------------------------------------
// Implicit-GEMM 3x3 conv, pad=1, stride=1, LDS-staged activations.
// Block = one M-tile (16 x-contiguous pixels). 256 threads cooperatively fill
// a zero-padded [3 rows][18 px][CIN] bf16 slab into LDS, then 8 waves each own
// up to two 16x16 N-tiles (dual accumulators): every LDS A fragment is loaded
// once and feeds two independent V_WMMA_F32_16X16X32_BF16 ops.
// Unroll caps keep live ranges < 128 VGPRs (no spills, no vgpr_msb).
// EPI: 0 = BN+lrelu -> bf16 NHWC   1 = +bias, clip ±16 -> f32 NHWC
//      2 = BN        -> bf16 NHWC  3 = BN              -> f32 NHWC
//      4 = BN+lrelu -> f32 NCHW (Co=256, final output)
// ---------------------------------------------------------------------------
template<int CIN, int NPAD, int EPI>
__global__ __launch_bounds__(256) void conv_wmma(
    const __bf16* __restrict__ src, int srcStride, int srcChOff,
    const __bf16* __restrict__ wT,
    const float* __restrict__ scv, const float* __restrict__ shv, int realN,
    void* __restrict__ outp, int outStride, int outChOff)
{
    constexpr int NT = NPAD / 16;
    constexpr int NTPW = (NT + 7) / 8;         // N-tiles per wave (1 or 2)
    extern __shared__ char smem_raw[];
    __bf16* As = (__bf16*)smem_raw;            // [3][18][CIN], zero-padded halo

    int mtile = blockIdx.x;
    int p0 = mtile * 16;
    int b = p0 >> 12, rem = p0 & (kHW - 1);
    int y = rem >> 6, x0 = rem & (kW - 1);

    // ---- cooperative fill: 54 pixels x CIN channels, b128 granules --------
    constexpr int NVEC = 54 * CIN / 8;
    for (int i = threadIdx.x; i < NVEC; i += 256) {
        int e  = i * 8;
        int c  = e % CIN;          // multiple of 8
        int pl = e / CIN;          // 0..53
        int row = pl / 18, px = pl - row * 18;
        int yy = y + row - 1, xx = x0 + px - 1;
        uint4 val = make_uint4(0u, 0u, 0u, 0u);
        if (yy >= 0 && yy < kH && xx >= 0 && xx < kW)
            val = *(const uint4*)(src +
                   ((size_t)((b << 12) + yy * kW + xx)) * srcStride + srcChOff + c);
        *(uint4*)(As + (size_t)pl * CIN + c) = val;
    }
    __syncthreads();

    int wave = threadIdx.x >> 5, lane = threadIdx.x & 31;
    int m = lane & 15;
    int n = lane & 15;
    int kbase = (lane >> 4) ? 8 : 0;

    v8f zero = {};
    v8f acc[NTPW];
    int co[NTPW];
    bool act[NTPW];
    #pragma unroll
    for (int u = 0; u < NTPW; ++u) {
        int nt = wave + u * 8;
        act[u] = (nt < NT);
        co[u] = (act[u] ? nt : 0) * 16 + n;
        acc[u] = zero;
    }

    if (wave < NT) {
        #pragma unroll 1
        for (int t = 0; t < 9; ++t) {
            int row = t / 3, kx = t % 3;       // LDS slab is already padded
            const __bf16* ap  = As + ((size_t)(row * 18 + m + kx)) * CIN + kbase;
            const __bf16* bpt = wT + ((size_t)t * NPAD) * CIN + kbase;
            __builtin_prefetch(bpt + (size_t)co[0] * CIN, 0, 1);  // global_prefetch_b8
            #pragma unroll 2
            for (int kc = 0; kc < CIN / 32; ++kc) {
                Frag A;
                A.q[0] = *(const uint4*)(ap + kc * 32);           // ds_load_b128
                A.q[1] = *(const uint4*)(ap + kc * 32 + 16);
                #pragma unroll
                for (int u = 0; u < NTPW; ++u) {
                    if (!act[u]) continue;                        // wave-uniform
                    const __bf16* bp = bpt + (size_t)co[u] * CIN;
                    Frag Bm;
                    Bm.q[0] = *(const uint4*)(bp + kc * 32);      // global b128
                    Bm.q[1] = *(const uint4*)(bp + kc * 32 + 16);
                    acc[u] = __builtin_amdgcn_wmma_f32_16x16x32_bf16(
                        false, A.v, false, Bm.v, (short)0, acc[u], false, false);
                }
            }
        }
    }

    // ---- epilogue: lane holds col n=co[u], rows mm = r + 8*(lane>=16) -----
    #pragma unroll
    for (int u = 0; u < NTPW; ++u) {
        if (!act[u]) continue;
        int c = co[u];
        float s = 1.f, h = 0.f;
        if (EPI == 1) {
            h = (c < realN) ? shv[c] : 0.f;            // bias only
        } else {
            if (c < realN) { s = scv[c]; h = shv[c]; }
            else           { s = 0.f;    h = 0.f;     }
        }
        #pragma unroll
        for (int r = 0; r < 8; ++r) {
            int mm = r + ((lane >> 4) << 3);
            int px = p0 + mm;
            float v = acc[u][r];
            if (EPI == 1) {
                v = v + h;
                v = fminf(fmaxf(v, -16.f), 16.f);      // max_offset = 64/4
                ((float*)outp)[(size_t)px * outStride + outChOff + c] = v;
            } else if (EPI == 0) {
                v = v * s + h;
                v = (v > 0.f) ? v : 0.1f * v;
                ((__bf16*)outp)[(size_t)px * outStride + outChOff + c] = (__bf16)v;
            } else if (EPI == 2) {
                v = v * s + h;
                ((__bf16*)outp)[(size_t)px * outStride + outChOff + c] = (__bf16)v;
            } else if (EPI == 3) {
                v = v * s + h;
                ((float*)outp)[(size_t)px * outStride + outChOff + c] = v;
            } else { // EPI == 4: fp32 NCHW final output
                v = v * s + h;
                v = (v > 0.f) ? v : 0.1f * v;
                int bb = px >> 12, sp = px & (kHW - 1);
                ((float*)outp)[((size_t)bb * 256 + c) * kHW + sp] = v;
            }
        }
    }
}

// ---------------------------------------------------------------------------
// Deformable 3x3 conv, LDS-staged sampled patches.
// Phase 1: 256 threads cooperatively bilinear-sample [9 taps][16 px][256 ci]
// into LDS (4 corner b128 loads + fma blend per 8-channel group).
// Phase 2: 8 waves x 2 N-tiles (dual accumulators), A from LDS once per
// fragment, two WMMAs per A. Output (+bias) -> X_align slot of cat2.
// ---------------------------------------------------------------------------
__global__ __launch_bounds__(256) void deform_wmma(
    const __bf16* __restrict__ cat1, const float* __restrict__ offs,
    const __bf16* __restrict__ wT, const float* __restrict__ bias,
    __bf16* __restrict__ cat2)
{
    constexpr int CIN = 256, NPAD = 256;
    extern __shared__ char smem_raw[];
    __bf16* As = (__bf16*)smem_raw;            // [9][16][256]

    int mtile = blockIdx.x;
    int p0 = mtile * 16;
    int b = p0 >> 12, rem = p0 & (kHW - 1);
    int y = rem >> 6, x0 = rem & (kW - 1);

    // ---- phase 1: cooperative bilinear sampling (9*16*32 = 4608 tasks) ----
    #pragma unroll 1
    for (int i = threadIdx.x; i < 9 * 16 * 32; i += 256) {
        int cg = i & 31;                 // channel group (8 ch)
        int pm = (i >> 5) & 15;          // pixel in tile
        int t  = i >> 9;                 // tap 0..8
        int c0 = cg * 8;
        int p  = p0 + pm;
        int x  = x0 + pm;
        float dy = offs[(size_t)p * 32 + 2 * t + 0];
        float dx = offs[(size_t)p * 32 + 2 * t + 1];
        float ys = (float)(y + t / 3 - 1) + dy;
        float xs = (float)(x + t % 3 - 1) + dx;
        float fy0 = floorf(ys), fx0 = floorf(xs);
        int iy0 = (int)fy0, ix0 = (int)fx0;
        float fy = ys - fy0, fx = xs - fx0;
        float w00 = (1.f - fy) * (1.f - fx), w01 = (1.f - fy) * fx;
        float w10 = fy * (1.f - fx),          w11 = fy * fx;
        bool vy0 = (iy0 >= 0) && (iy0 < kH);
        bool vy1 = (iy0 + 1 >= 0) && (iy0 + 1 < kH);
        bool vx0 = (ix0 >= 0) && (ix0 < kW);
        bool vx1 = (ix0 + 1 >= 0) && (ix0 + 1 < kW);
        long base00 = ((long)(b << 12) + (long)iy0 * kW + ix0) * 512 + 256 + c0;
        long base01 = base00 + 512;
        long base10 = base00 + (long)kW * 512;
        long base11 = base10 + 512;
        uint4 z = make_uint4(0u, 0u, 0u, 0u);
        H8 q00, q01, q10, q11, o;
        q00.q = (vy0 && vx0) ? *(const uint4*)(cat1 + base00) : z;
        q01.q = (vy0 && vx1) ? *(const uint4*)(cat1 + base01) : z;
        q10.q = (vy1 && vx0) ? *(const uint4*)(cat1 + base10) : z;
        q11.q = (vy1 && vx1) ? *(const uint4*)(cat1 + base11) : z;
        #pragma unroll
        for (int j = 0; j < 8; ++j) {
            float sv = w00 * (float)q00.h[j] + w01 * (float)q01.h[j]
                     + w10 * (float)q10.h[j] + w11 * (float)q11.h[j];
            o.h[j] = (__bf16)sv;
        }
        *(uint4*)(As + ((size_t)(t * 16 + pm)) * CIN + c0) = o.q;
    }
    __syncthreads();

    // ---- phase 2: WMMA GEMM from LDS patches, 2 N-tiles per wave ----------
    int wave = threadIdx.x >> 5, lane = threadIdx.x & 31;
    int m = lane & 15;
    int kbase = (lane >> 4) ? 8 : 0;
    int co0 = wave * 16 + (lane & 15);
    int co1 = co0 + 128;
    v8f acc0 = {}, acc1 = {};
    #pragma unroll 1
    for (int t = 0; t < 9; ++t) {
        const __bf16* ap  = As + ((size_t)(t * 16 + m)) * CIN + kbase;
        const __bf16* bp0 = wT + ((size_t)(t * NPAD + co0)) * CIN + kbase;
        const __bf16* bp1 = wT + ((size_t)(t * NPAD + co1)) * CIN + kbase;
        __builtin_prefetch(bp0, 0, 1);
        #pragma unroll 2
        for (int kc = 0; kc < CIN / 32; ++kc) {
            Frag A, B0, B1;
            A.q[0]  = *(const uint4*)(ap + kc * 32);
            A.q[1]  = *(const uint4*)(ap + kc * 32 + 16);
            B0.q[0] = *(const uint4*)(bp0 + kc * 32);
            B0.q[1] = *(const uint4*)(bp0 + kc * 32 + 16);
            B1.q[0] = *(const uint4*)(bp1 + kc * 32);
            B1.q[1] = *(const uint4*)(bp1 + kc * 32 + 16);
            acc0 = __builtin_amdgcn_wmma_f32_16x16x32_bf16(
                false, A.v, false, B0.v, (short)0, acc0, false, false);
            acc1 = __builtin_amdgcn_wmma_f32_16x16x32_bf16(
                false, A.v, false, B1.v, (short)0, acc1, false, false);
        }
    }
    float b0 = bias[co0], b1 = bias[co1];
    #pragma unroll
    for (int r = 0; r < 8; ++r) {
        int mm = r + ((lane >> 4) << 3);
        int px = p0 + mm;
        cat2[(size_t)px * 768 + 256 + co0] = (__bf16)(acc0[r] + b0);
        cat2[(size_t)px * 768 + 256 + co1] = (__bf16)(acc1[r] + b1);
    }
}

// ---------------------------------------------------------------------------
// Softmax over 9 channels (stride-16 f32 rows)
// ---------------------------------------------------------------------------
__global__ void softmax9_kernel(const float* __restrict__ e2, float* __restrict__ kb)
{
    int pIdx = blockIdx.x * blockDim.x + threadIdx.x;
    if (pIdx >= kNPix) return;
    const float* r = e2 + (size_t)pIdx * 16;
    float mx = r[0];
    #pragma unroll
    for (int k = 1; k < 9; ++k) mx = fmaxf(mx, r[k]);
    float e[9], s = 0.f;
    #pragma unroll
    for (int k = 0; k < 9; ++k) { e[k] = __expf(r[k] - mx); s += e[k]; }
    float inv = 1.f / s;
    #pragma unroll
    for (int k = 0; k < 9; ++k) kb[(size_t)pIdx * 16 + k] = e[k] * inv;
}

// ---------------------------------------------------------------------------
// X_down = sum_k kern[k] * unfold3x3(X_align)[k]  (reads/writes cat2 slots)
// One thread per (pixel, 8-channel chunk), b128 neighbor loads.
// ---------------------------------------------------------------------------
__global__ void xdown_kernel(const float* __restrict__ kb,
                             const __bf16* __restrict__ cat2in,
                             __bf16* __restrict__ cat2out)
{
    int idx = blockIdx.x * blockDim.x + threadIdx.x;
    if (idx >= kNPix * 32) return;
    int p = idx >> 5, cg = idx & 31;
    int c0 = cg * 8;
    int b = p >> 12, rem = p & (kHW - 1);
    int y = rem >> 6, x = rem & (kW - 1);
    float acc[8] = {0.f, 0.f, 0.f, 0.f, 0.f, 0.f, 0.f, 0.f};
    for (int t = 0; t < 9; ++t) {
        int ny = y + t / 3 - 1, nx = x + t % 3 - 1;
        if (ny < 0 || ny >= kH || nx < 0 || nx >= kW) continue;  // zero pad
        float kwgt = kb[(size_t)p * 16 + t];
        H8 q;
        q.q = *(const uint4*)(cat2in +
              ((size_t)((b << 12) + ny * kW + nx)) * 768 + 256 + c0);
        #pragma unroll
        for (int j = 0; j < 8; ++j) acc[j] += kwgt * (float)q.h[j];
    }
    H8 o;
    #pragma unroll
    for (int j = 0; j < 8; ++j) o.h[j] = (__bf16)acc[j];
    *(uint4*)(cat2out + (size_t)p * 768 + 512 + c0) = o.q;  // X_down slot
}

// ---------------------------------------------------------------------------
extern "C" void kernel_launch(void* const* d_in, const int* in_sizes, int n_in,
                              void* d_out, int out_size, void* d_ws, size_t ws_size,
                              hipStream_t stream)
{
    const float* X_O   = (const float*)d_in[0];
    const float* X_in  = (const float*)d_in[1];
    const float* cf_w  = (const float*)d_in[2];
    const float* cf_s  = (const float*)d_in[3];
    const float* cf_sh = (const float*)d_in[4];
    const float* off_w = (const float*)d_in[5];
    const float* off_b = (const float*)d_in[6];
    const float* al_w  = (const float*)d_in[7];
    const float* al_b  = (const float*)d_in[8];
    const float* e1_w  = (const float*)d_in[9];
    const float* e1_s  = (const float*)d_in[10];
    const float* e1_sh = (const float*)d_in[11];
    const float* e2_w  = (const float*)d_in[12];
    const float* e2_s  = (const float*)d_in[13];
    const float* e2_sh = (const float*)d_in[14];
    const float* fo_w  = (const float*)d_in[15];
    const float* fo_s  = (const float*)d_in[16];
    const float* fo_sh = (const float*)d_in[17];
    (void)in_sizes; (void)n_in; (void)out_size; (void)ws_size;

    char* wsp = (char*)d_ws;
    auto alloc = [&](size_t bytes) -> char* {
        char* q = wsp;
        wsp += (bytes + 255) & ~(size_t)255;
        return q;
    };
    // activations (NHWC bf16 / f32); concat buffers shared by producers
    __bf16* cat1 = (__bf16*)alloc((size_t)kNPix * 512 * 2); // [X_O | X_in]
    __bf16* cat2 = (__bf16*)alloc((size_t)kNPix * 768 * 2); // [X_off | X_align | X_down]
    float*  offs = (float*) alloc((size_t)kNPix * 32 * 4);  // 18 used, pad 32
    __bf16* e1o  = (__bf16*)alloc((size_t)kNPix * 64 * 2);
    float*  e2o  = (float*) alloc((size_t)kNPix * 16 * 4);  // 9 used, pad 16
    float*  kb   = (float*) alloc((size_t)kNPix * 16 * 4);
    // packed bf16 weights wT[tap][co_pad][ci]
    __bf16* wcf = (__bf16*)alloc((size_t)9 * 256 * 512 * 2);
    __bf16* wof = (__bf16*)alloc((size_t)9 * 32  * 256 * 2);
    __bf16* wal = (__bf16*)alloc((size_t)9 * 256 * 256 * 2);
    __bf16* we1 = (__bf16*)alloc((size_t)9 * 64  * 256 * 2);
    __bf16* we2 = (__bf16*)alloc((size_t)9 * 16  * 64  * 2);
    __bf16* wfo = (__bf16*)alloc((size_t)9 * 256 * 768 * 2);

    auto cdiv = [](int a, int b) { return (a + b - 1) / b; };

    // 1) pack weights
    pack_w_kernel<<<cdiv(9*256*512, 256), 256, 0, stream>>>(cf_w, wcf, 256, 512, 256);
    pack_w_kernel<<<cdiv(9*32*256,  256), 256, 0, stream>>>(off_w, wof, 18, 256, 32);
    pack_w_kernel<<<cdiv(9*256*256, 256), 256, 0, stream>>>(al_w, wal, 256, 256, 256);
    pack_w_kernel<<<cdiv(9*64*256,  256), 256, 0, stream>>>(e1_w, we1, 64, 256, 64);
    pack_w_kernel<<<cdiv(9*16*64,   256), 256, 0, stream>>>(e2_w, we2, 9, 64, 16);
    pack_w_kernel<<<cdiv(9*256*768, 256), 256, 0, stream>>>(fo_w, wfo, 256, 768, 256);

    // 2) fp32 NCHW inputs -> bf16 NHWC concat
    nchw_to_slot_kernel<<<cdiv(kNPix*256, 256), 256, 0, stream>>>(X_O,  cat1, 256, 512, 0);
    nchw_to_slot_kernel<<<cdiv(kNPix*256, 256), 256, 0, stream>>>(X_in, cat1, 256, 512, 256);

    // 3) cat_fuse conv (BN+lrelu) -> X_off (cat2 slot 0)
    conv_wmma<512, 256, 0><<<kMTiles, 256, 54*512*2, stream>>>(
        cat1, 512, 0, wcf, cf_s, cf_sh, 256, cat2, 768, 0);
    // 4) offset conv (+bias, clip ±16) -> offs f32
    conv_wmma<256, 32, 1><<<kMTiles, 256, 54*256*2, stream>>>(
        cat2, 768, 0, wof, nullptr, off_b, 18, offs, 32, 0);
    // 5) enc1 (BN) -> e1o ; enc2 (BN) -> e2o
    conv_wmma<256, 64, 2><<<kMTiles, 256, 54*256*2, stream>>>(
        cat2, 768, 0, we1, e1_s, e1_sh, 64, e1o, 64, 0);
    conv_wmma<64, 16, 3><<<kMTiles, 256, 54*64*2, stream>>>(
        e1o, 64, 0, we2, e2_s, e2_sh, 9, e2o, 16, 0);
    // 6) deformable conv -> X_align (cat2 slot 256)
    deform_wmma<<<kMTiles, 256, 9*16*256*2, stream>>>(cat1, offs, wal, al_b, cat2);
    // 7) softmax over 9 taps; 8) kern (x) unfold(X_align) -> X_down (cat2 slot 512)
    softmax9_kernel<<<cdiv(kNPix, 256), 256, 0, stream>>>(e2o, kb);
    xdown_kernel<<<cdiv(kNPix*32, 256), 256, 0, stream>>>(kb, cat2, cat2);
    // 9) fuse_out conv (BN+lrelu) -> fp32 NCHW d_out
    conv_wmma<768, 256, 4><<<kMTiles, 256, 54*768*2, stream>>>(
        cat2, 768, 0, wfo, fo_s, fo_sh, 256, d_out, 0, 0);
}
